// SpikingEmbed_79216376807560
// MI455X (gfx1250) — matrix operand order repaired and checked by reading, not compiled, not run on hardware
//
#include <hip/hip_runtime.h>

// Shapes from the reference
#define T_STEPS 4
#define BATCH   8
#define CIN     128
#define NLEN    4096
#define COUT    256

#define NT_PER_WAVE 4                           // n-tiles (of 16) per wave
#define GBLOCKS (BATCH * (NLEN / 16 / NT_PER_WAVE))  // 512 blocks per GEMM pass
#define NCOUNT  ((float)(T_STEPS * BATCH * NLEN))    // 131072 per channel

typedef __attribute__((ext_vector_type(16))) __bf16 v16bf;
typedef __attribute__((ext_vector_type(8)))  __bf16 v8bf;
typedef __attribute__((ext_vector_type(8)))  float  v8f;

// f32 -> bf16 round-to-nearest-even (bit ops; only in the two tiny prep passes)
static __device__ __forceinline__ __bf16 f2bf(float f) {
  unsigned u = __builtin_bit_cast(unsigned, f);
  u += 0x7FFFu + ((u >> 16) & 1u);
  unsigned short s = (unsigned short)(u >> 16);
  return __builtin_bit_cast(__bf16, s);
}

static __device__ __forceinline__ v16bf cat8(v8bf lo, v8bf hi) {
  return __builtin_shufflevector(lo, hi, 0, 1, 2, 3, 4, 5, 6, 7,
                                 8, 9, 10, 11, 12, 13, 14, 15);
}

// ---------------------------------------------------------------------------
// Pass 0a: x [TB][Cin][N] f32 -> Xt [TB][N][Cin] bf16 (transpose + convert).
// ---------------------------------------------------------------------------
__global__ __launch_bounds__(256) void k_transpose_bf16(const float* __restrict__ X,
                                                        __bf16* __restrict__ Xt) {
  __shared__ __bf16 tile[32][CIN + 8];       // [n][c], +8 pad keeps rows 16B-aligned
  const int tb = blockIdx.x >> 7;            // 128 n-chunks of 32 per (t,b)
  const int n0 = (blockIdx.x & 127) * 32;
  const int tid = threadIdx.x;

  const float* src = X + (size_t)tb * CIN * NLEN + n0;
#pragma unroll
  for (int i = 0; i < 16; ++i) {
    int e = i * 256 + tid;
    int c = e >> 5, n = e & 31;              // consecutive tid -> consecutive n
    tile[n][c] = f2bf(src[(size_t)c * NLEN + n]);
  }
  __syncthreads();

  const int n  = tid >> 3;
  const int cg = (tid & 7) * 16;
  const uint4* lp = reinterpret_cast<const uint4*>(&tile[n][cg]);
  uint4 w0 = lp[0];
  uint4 w1 = lp[1];
  uint4* dst = reinterpret_cast<uint4*>(Xt + ((size_t)tb * NLEN + n0 + n) * CIN + cg);
  dst[0] = w0;
  dst[1] = w1;
}

// ---------------------------------------------------------------------------
// Pass 0b: pre-pack W (f32) into bf16 WMMA A-fragment layout.
// Wp[otile][kb][lane][16]: lane m=lane&15 is row; lanes<16 hold
// K={0..15}+32*kb, lanes>=16 hold K={16..31}+32*kb.
// ---------------------------------------------------------------------------
__global__ __launch_bounds__(256) void k_pack_w(const float* __restrict__ W,
                                                __bf16* __restrict__ Wp) {
  const int idx  = blockIdx.x * 256 + threadIdx.x;  // 0..2047
  const int lane = idx & 31;
  const int kb   = (idx >> 5) & 3;
  const int ot   = idx >> 7;
  const int m = lane & 15;
  const int g = lane >> 4;
  const float* wrow = W + (ot * 16 + m) * CIN;
  __bf16* dst = Wp + (size_t)idx * 16;
#pragma unroll
  for (int j = 0; j < 8; ++j) {
    dst[j]     = f2bf(wrow[kb * 32 + g * 16 + j]);
    dst[8 + j] = f2bf(wrow[kb * 32 + g * 16 + 8 + j]);
  }
}

// A fragments from packed layout: 2 coalesced b128 loads per kb, zero VALU.
static __device__ __forceinline__ void load_a_packed(const __bf16* __restrict__ Wp,
                                                     int otile, int lane, v16bf a[4]) {
#pragma unroll
  for (int kb = 0; kb < 4; ++kb) {
    const v8bf* p = reinterpret_cast<const v8bf*>(
        Wp + (((size_t)otile * 4 + kb) * 32 + lane) * 16);
    a[kb] = cat8(p[0], p[1]);
  }
}

// B fragment: 16 consecutive bf16 from transposed tensor = 2x b128 loads.
static __device__ __forceinline__ v16bf load_b_frag(const __bf16* __restrict__ xrow,
                                                    int kb) {
  const v8bf* p = reinterpret_cast<const v8bf*>(xrow + kb * 32);
  return cat8(p[0], p[1]);
}

// ---------------------------------------------------------------------------
// Pass 1: GEMM (M=32, N=64 per wave) + per-channel {sum, sumsq} partials.
// partials: [GBLOCKS][COUT][2] floats. Deterministic.
// ---------------------------------------------------------------------------
__global__ __launch_bounds__(256) void k_gemm_stats(const __bf16* __restrict__ Xt,
                                                    const __bf16* __restrict__ Wp,
                                                    float* __restrict__ partials) {
  const int lane = threadIdx.x & 31;
  const int wave = threadIdx.x >> 5;         // 8 waves: o-slices of 32
  const int b  = blockIdx.x >> 6;            // 64 n-chunks of 64 per batch
  const int n0 = (blockIdx.x & 63) * 64;
  const int o0 = wave * 32;
  const int g  = lane >> 4;

  v16bf a0[4], a1[4];
  load_a_packed(Wp, wave * 2,     lane, a0);
  load_a_packed(Wp, wave * 2 + 1, lane, a1);

  float s1[16], s2[16];
#pragma unroll
  for (int r = 0; r < 16; ++r) { s1[r] = 0.f; s2[r] = 0.f; }

#pragma unroll 1   // real loop: keep VGPRs < 256, avoid MSB-mode churn
  for (int nt = 0; nt < NT_PER_WAVE; ++nt) {
#pragma unroll
    for (int t = 0; t < T_STEPS; ++t) {
      const __bf16* xrow =
          Xt + ((size_t)(t * BATCH + b) * NLEN + n0 + nt * 16 + (lane & 15)) * CIN
             + g * 16;
      v8f acc0 = {0.f, 0.f, 0.f, 0.f, 0.f, 0.f, 0.f, 0.f};
      v8f acc1 = {0.f, 0.f, 0.f, 0.f, 0.f, 0.f, 0.f, 0.f};
#pragma unroll
      for (int kb = 0; kb < 4; ++kb) {
        v16bf bf = load_b_frag(xrow, kb);
        acc0 = __builtin_amdgcn_wmma_f32_16x16x32_bf16(false, a0[kb], false, bf,
                                                       (short)0, acc0, false, false);
        acc1 = __builtin_amdgcn_wmma_f32_16x16x32_bf16(false, a1[kb], false, bf,
                                                       (short)0, acc1, false, false);
      }
#pragma unroll
      for (int r = 0; r < 8; ++r) {
        float y0 = acc0[r], y1 = acc1[r];
        s1[r]     += y0;  s2[r]     += y0 * y0;
        s1[8 + r] += y1;  s2[8 + r] += y1 * y1;
      }
    }
  }

  // Reduce across the 16 lanes sharing each output row (xor stays in-group)
#pragma unroll
  for (int r = 0; r < 16; ++r) {
#pragma unroll
    for (int mask = 1; mask < 16; mask <<= 1) {
      s1[r] += __shfl_xor(s1[r], mask, 32);
      s2[r] += __shfl_xor(s2[r], mask, 32);
    }
  }
  if ((lane & 15) == 0) {                    // lanes 0/16 hold rows g*8 + r
    float* p0 = partials + ((size_t)blockIdx.x * COUT + o0 + g * 8) * 2;
    float* p1 = partials + ((size_t)blockIdx.x * COUT + o0 + 16 + g * 8) * 2;
#pragma unroll
    for (int r = 0; r < 8; ++r) {
      p0[2 * r] = s1[r];      p0[2 * r + 1] = s2[r];
      p1[2 * r] = s1[8 + r];  p1[2 * r + 1] = s2[8 + r];
    }
  }
}

// Pass 1b: fold partials -> per-channel scale/shift (BN affine folded in).
__global__ __launch_bounds__(256) void k_reduce_stats(const float* __restrict__ partials,
                                                      const float* __restrict__ gamma,
                                                      const float* __restrict__ beta,
                                                      float* __restrict__ scaleshift) {
  const int o = threadIdx.x;
  const float2* p = (const float2*)partials;
  float s1 = 0.f, s2 = 0.f;
  for (int gblk = 0; gblk < GBLOCKS; ++gblk) {
    float2 v = p[(size_t)gblk * COUT + o];
    s1 += v.x;
    s2 += v.y;
  }
  const float inv = 1.0f / NCOUNT;
  float mean = s1 * inv;
  float var  = fmaxf(s2 * inv - mean * mean, 0.0f);
  float sc = gamma[o] * rsqrtf(var + 1e-5f);
  float sh = beta[o] - mean * sc;
  scaleshift[2 * o]     = sc;
  scaleshift[2 * o + 1] = sh;
}

// ---------------------------------------------------------------------------
// Pass 2: GEMM again (Xt is L2-resident) + BN + 4-step LIF, emit spikes.
// ---------------------------------------------------------------------------
__global__ __launch_bounds__(256) void k_gemm_lif(const __bf16* __restrict__ Xt,
                                                  const __bf16* __restrict__ Wp,
                                                  const float* __restrict__ scaleshift,
                                                  float* __restrict__ out) {
  const int lane = threadIdx.x & 31;
  const int wave = threadIdx.x >> 5;
  const int b  = blockIdx.x >> 6;
  const int n0 = (blockIdx.x & 63) * 64;
  const int o0 = wave * 32;
  const int g  = lane >> 4;

  v16bf a0[4], a1[4];
  load_a_packed(Wp, wave * 2,     lane, a0);
  load_a_packed(Wp, wave * 2 + 1, lane, a1);

  float sc[16], sh[16];
#pragma unroll
  for (int r = 0; r < 8; ++r) {
    int c0 = o0 + g * 8 + r;
    int c1 = o0 + 16 + g * 8 + r;
    sc[r]     = scaleshift[2 * c0];  sh[r]     = scaleshift[2 * c0 + 1];
    sc[8 + r] = scaleshift[2 * c1];  sh[8 + r] = scaleshift[2 * c1 + 1];
  }

#pragma unroll 1   // real loop: keep VGPRs < 256, avoid MSB-mode churn
  for (int nt = 0; nt < NT_PER_WAVE; ++nt) {
    const int n = n0 + nt * 16 + (lane & 15);
    float v[16];
#pragma unroll
    for (int r = 0; r < 16; ++r) v[r] = 0.f;   // LIF membrane per n-tile

#pragma unroll
    for (int t = 0; t < T_STEPS; ++t) {
      const __bf16* xrow =
          Xt + ((size_t)(t * BATCH + b) * NLEN + n) * CIN + g * 16;
      v8f acc0 = {0.f, 0.f, 0.f, 0.f, 0.f, 0.f, 0.f, 0.f};
      v8f acc1 = {0.f, 0.f, 0.f, 0.f, 0.f, 0.f, 0.f, 0.f};
#pragma unroll
      for (int kb = 0; kb < 4; ++kb) {
        v16bf bf = load_b_frag(xrow, kb);
        acc0 = __builtin_amdgcn_wmma_f32_16x16x32_bf16(false, a0[kb], false, bf,
                                                       (short)0, acc0, false, false);
        acc1 = __builtin_amdgcn_wmma_f32_16x16x32_bf16(false, a1[kb], false, bf,
                                                       (short)0, acc1, false, false);
      }
      float* orow = out + ((size_t)(t * BATCH + b) * COUT) * NLEN + n;
#pragma unroll
      for (int r = 0; r < 8; ++r) {
        {
          float y  = acc0[r] * sc[r] + sh[r];        // batch-norm
          float vn = 0.5f * (v[r] + y);              // tau = 2
          float s  = (vn >= 0.8f) ? 1.0f : 0.0f;     // Heaviside(v - vth)
          v[r] = vn * (1.0f - s);                    // hard reset
          orow[(size_t)(o0 + g * 8 + r) * NLEN] = s;
        }
        {
          float y  = acc1[r] * sc[8 + r] + sh[8 + r];
          float vn = 0.5f * (v[8 + r] + y);
          float s  = (vn >= 0.8f) ? 1.0f : 0.0f;
          v[8 + r] = vn * (1.0f - s);
          orow[(size_t)(o0 + 16 + g * 8 + r) * NLEN] = s;
        }
      }
    }
  }
}

extern "C" void kernel_launch(void* const* d_in, const int* in_sizes, int n_in,
                              void* d_out, int out_size, void* d_ws, size_t ws_size,
                              hipStream_t stream) {
  const float* x     = (const float*)d_in[0];  // [T,B,Cin,N] f32
  const float* w     = (const float*)d_in[1];  // [Cout,Cin] f32
  const float* gamma = (const float*)d_in[2];  // [Cout]
  const float* beta  = (const float*)d_in[3];  // [Cout]
  float* out = (float*)d_out;                  // [T,B,Cout,N] f32

  // Workspace: Xt (32 MiB) | partials (1 MiB) | scaleshift (2 KiB) | Wp (64 KiB)
  __bf16* Xt = (__bf16*)d_ws;
  float* partials   = (float*)((char*)d_ws + (size_t)T_STEPS * BATCH * CIN * NLEN * 2);
  float* scaleshift = partials + (size_t)GBLOCKS * COUT * 2;
  __bf16* Wp = (__bf16*)(scaleshift + 2 * COUT);

  k_transpose_bf16<<<dim3(T_STEPS * BATCH * (NLEN / 32)), dim3(256), 0, stream>>>(x, Xt);
  k_pack_w<<<dim3(8), dim3(256), 0, stream>>>(w, Wp);
  k_gemm_stats<<<dim3(GBLOCKS), dim3(256), 0, stream>>>(Xt, Wp, partials);
  k_reduce_stats<<<dim3(1), dim3(COUT), 0, stream>>>(partials, gamma, beta, scaleshift);
  k_gemm_lif<<<dim3(GBLOCKS), dim3(256), 0, stream>>>(Xt, Wp, scaleshift, out);
}
